// Encoder_22093311771091
// MI455X (gfx1250) — matrix operand order repaired and checked by reading, not compiled, not run on hardware
//
#include <hip/hip_runtime.h>

typedef __attribute__((ext_vector_type(16))) __bf16 v16bf;
typedef __attribute__((ext_vector_type(8)))  __bf16 v8bf;
typedef __attribute__((ext_vector_type(8)))  float  v8f;

__device__ __forceinline__ float sigmoidf_(float x) { return 1.0f / (1.0f + __expf(-x)); }

// ---------------------------------------------------------------------------
// Convert input x (N,D,H,W,8) f32 -> (N,D,H,W,32) bf16 zero-padded channels
// ---------------------------------------------------------------------------
__global__ __launch_bounds__(256) void convert_x_kernel(const float* __restrict__ x,
                                                        __bf16* __restrict__ xb,
                                                        size_t tot) {
    size_t idx = (size_t)blockIdx.x * blockDim.x + threadIdx.x;
    if (idx >= tot) return;
    int c = (int)(idx & 31);
    size_t v = idx >> 5;
    float val = (c < 8) ? x[v * 8 + c] : 0.0f;
    xb[idx] = (__bf16)val;
}

// ---------------------------------------------------------------------------
// Pre-swizzle conv weights (3,3,3,CIN,120) f32 into WMMA B-fragment layout:
//   wf[tap][chunk][ntile][lane][e]  (bf16)
//   B-matrix 32x16 (KxN): lane holds column N=lane&15, element e = row K (table order).
//   cin = chunk*32 + (e<8 ? 8*(lane>>4)+e : 16+8*(lane>>4)+e-8) ; cout = ntile*16 + (lane&15)
// ---------------------------------------------------------------------------
__global__ __launch_bounds__(256) void convert_wfrag_kernel(const float* __restrict__ w,
                                                            __bf16* __restrict__ wf,
                                                            int CIN, int NCHUNK, int COUT,
                                                            size_t tot) {
    size_t idx = (size_t)blockIdx.x * blockDim.x + threadIdx.x;
    if (idx >= tot) return;
    int e    = (int)(idx & 15);
    size_t t = idx >> 4;
    int lane = (int)(t & 31); t >>= 5;
    int nt   = (int)(t & 7);  t >>= 3;
    int ch   = (int)(t % NCHUNK);
    int tap  = (int)(t / NCHUNK);
    int cin  = ch * 32 + ((e < 8) ? (8 * (lane >> 4) + e) : (16 + 8 * (lane >> 4) + (e - 8)));
    int co   = nt * 16 + (lane & 15);
    float v = 0.0f;
    if (cin < CIN && co < COUT) v = w[((size_t)tap * CIN + cin) * COUT + co];
    wf[idx] = (__bf16)v;
}

// ---------------------------------------------------------------------------
// Fused implicit-GEMM conv3d (SAME, 3x3x3) + sliced LayerNorm + gate.
// Block: 256 threads = 8 waves (wave32). Block computes 16 voxels (along W)
// x 120 (pad 128) output channels. Wave i owns channel tile [16i, 16i+16).
// A receptive field staged into LDS via GLOBAL_LOAD_ASYNC_TO_LDS_B128.
// ---------------------------------------------------------------------------
template <int CINP, int NCHUNK>
__global__ __launch_bounds__(256) void conv_ln_gate_kernel(
    const __bf16* __restrict__ in, const __bf16* __restrict__ wfrag,
    const float* __restrict__ gamma, const float* __restrict__ beta,
    const float* __restrict__ gw,  // (32,24)
    float* __restrict__ out, int D, int H, int W) {
    const int W16 = W >> 4;
    int blk = blockIdx.x;
    int w16 = blk % W16; int t = blk / W16;
    int h = t % H; t /= H;
    int d = t % D; int n = t / D;
    const int w_base = w16 * 16;

    constexpr int ROWS = 3 * 3 * 18;  // receptive field rows for 16 voxels along W
    __shared__ __align__(16) __bf16 As[ROWS * CINP];
    __shared__ __align__(16) float  Os[16 * 128];
    __shared__ __align__(16) float  Gs[16 * 24];

    // ---- async stage of the receptive field into LDS (16B granules) ----
    constexpr int VPR  = CINP / 8;  // 16-byte vectors per row
    constexpr int TOTV = ROWS * VPR;
    for (int idx = threadIdx.x; idx < TOTV; idx += blockDim.x) {
        int row = idx / VPR;
        int cv  = (idx - row * VPR) * 8;
        int wx  = row % 18;
        int r2  = row / 18;
        int kh  = r2 % 3, kd = r2 / 3;
        int zd = d + kd - 1, yh = h + kh - 1, xw = w_base + wx - 1;
        __bf16* dst = As + (size_t)row * CINP + cv;
        if (zd >= 0 && zd < D && yh >= 0 && yh < H && xw >= 0 && xw < W) {
            size_t g = ((((size_t)n * D + zd) * H + yh) * (size_t)W + xw) * CINP + cv;
            // LDS byte offset = low 32 bits of the generic LDS pointer
            unsigned lds_off = (unsigned)(uintptr_t)dst;
            unsigned long long gaddr = (unsigned long long)(uintptr_t)(in + g);
            asm volatile("global_load_async_to_lds_b128 %0, %1, off"
                         :
                         : "v"(lds_off), "v"(gaddr)
                         : "memory");
        } else {
            v8bf z = {};
            *(v8bf*)dst = z;  // SAME-padding zero fill (DS path)
        }
    }
    asm volatile("s_wait_asynccnt 0x0" ::: "memory");
    __syncthreads();

    const int lane = threadIdx.x & 31;
    const int wave = threadIdx.x >> 5;  // ntile 0..7
    const int g    = lane >> 4;
    const int m    = lane & 15;

    union Frag { struct { v8bf lo, hi; } p; v16bf v; };

    v8f acc = {};
    for (int tap = 0; tap < 27; ++tap) {
        const int kd = tap / 9, kh = (tap / 3) % 3, kw = tap % 3;
        // A layout (16-bit A 16x32): lane (g,m): e<8 -> K = e + 8g ; e>=8 -> K = e+8 + 8g
        const __bf16* arow = As + ((size_t)((kd * 3 + kh) * 18 + (m + kw))) * CINP + g * 8;
        const __bf16* wrow = wfrag + ((size_t)tap * NCHUNK) * 4096 + (size_t)(wave * 32 + lane) * 16;
#pragma unroll
        for (int c = 0; c < NCHUNK; ++c) {
            Frag a, b;
            a.p.lo = *(const v8bf*)(arow + c * 32);
            a.p.hi = *(const v8bf*)(arow + c * 32 + 16);
            b.p.lo = *(const v8bf*)(wrow + (size_t)c * 4096);
            b.p.hi = *(const v8bf*)(wrow + (size_t)c * 4096 + 8);
            acc = __builtin_amdgcn_wmma_f32_16x16x32_bf16(false, a.v, false, b.v,
                                                          (short)0, acc, false, false);
        }
    }

    // ---- spill C tile to LDS: lane N = lane&15, row r -> voxel r + 8*(lane>>4)
    {
        int co = wave * 16 + (lane & 15);
        int m0 = (lane >> 4) * 8;
        if (co < 120) {
#pragma unroll
            for (int r = 0; r < 8; ++r) Os[(m0 + r) * 128 + co] = acc[r];
        }
    }
    __syncthreads();

    const int tid = threadIdx.x;

    // ---- Phase A: sliced LayerNorm (one thread per voxel), normalize in place
    if (tid < 16) {
        float* xr = Os + tid * 128;
        const int sl[4] = {0, 32, 80, 120};
#pragma unroll
        for (int s = 0; s < 3; ++s) {
            int a0 = sl[s], a1 = sl[s + 1];
            float inv = 1.0f / (float)(a1 - a0);
            float mu = 0.0f;
            for (int c = a0; c < a1; ++c) mu += xr[c];
            mu *= inv;
            float var = 0.0f;
            for (int c = a0; c < a1; ++c) { float dd = xr[c] - mu; var += dd * dd; }
            var *= inv;
            float rs = __frsqrt_rn(var + 1e-5f);
            for (int c = a0; c < a1; ++c) xr[c] = (xr[c] - mu) * rs * gamma[c] + beta[c];
        }
    }
    __syncthreads();

    // ---- Phase B: gate GEMV 32->24 (8 threads per voxel, 3 gates each)
    if (tid < 128) {
        int v  = tid >> 3;
        int k8 = tid & 7;
        const float* xr = Os + v * 128;
#pragma unroll
        for (int jj = 0; jj < 3; ++jj) {
            int j = k8 * 3 + jj;
            float a2 = 0.0f;
            for (int i = 0; i < 32; ++i) a2 += xr[i] * gw[i * 24 + j];
            Gs[v * 24 + j] = sigmoidf_(a2);
        }
    }
    __syncthreads();

    // ---- Phase C: SiLU + gated writes (all 256 threads, 1920 elements)
    {
        size_t ob = ((((size_t)n * D + d) * H + h) * (size_t)W + w_base) * 120;
        for (int idx = tid; idx < 16 * 120; idx += 256) {
            int v = idx / 120;
            int c = idx - v * 120;
            float xv = Os[v * 128 + c];
            float r;
            if (c < 32) {
                r = xv * sigmoidf_(xv);
            } else {
                int gi = c - 32;
                int j = (gi < 48) ? (gi / 3) : (16 + (gi - 48) / 5);
                r = Gs[v * 24 + j] * xv;
            }
            out[ob + (size_t)v * 120 + c] = r;
        }
    }
}

// ---------------------------------------------------------------------------
// 2x2x2 average pool: (N,2Do,2Ho,2Wo,120) f32 -> (N,Do,Ho,Wo,128) bf16 padded
// ---------------------------------------------------------------------------
__global__ __launch_bounds__(256) void avgpool_to_bf16_kernel(const float* __restrict__ in,
                                                              __bf16* __restrict__ outp,
                                                              int Do, int Ho, int Wo, size_t tot) {
    size_t idx = (size_t)blockIdx.x * blockDim.x + threadIdx.x;
    if (idx >= tot) return;
    int c = (int)(idx & 127);
    size_t v = idx >> 7;
    int wo = (int)(v % Wo); v /= Wo;
    int ho = (int)(v % Ho); v /= Ho;
    int dd = (int)(v % Do);
    int n  = (int)(v / Do);
    float r = 0.0f;
    if (c < 120) {
        int Hi = 2 * Ho, Wi = 2 * Wo, Di = 2 * Do;
#pragma unroll
        for (int dz = 0; dz < 2; ++dz)
#pragma unroll
            for (int dy = 0; dy < 2; ++dy)
#pragma unroll
                for (int dx = 0; dx < 2; ++dx) {
                    size_t vv = (((size_t)n * Di + (2 * dd + dz)) * Hi + (2 * ho + dy)) * Wi +
                                (2 * wo + dx);
                    r += in[vv * 120 + c];
                }
        r *= 0.125f;
    }
    outp[idx] = (__bf16)r;
}

// ---------------------------------------------------------------------------
extern "C" void kernel_launch(void* const* d_in, const int* in_sizes, int n_in,
                              void* d_out, int out_size, void* d_ws, size_t ws_size,
                              hipStream_t stream) {
    (void)in_sizes; (void)n_in; (void)out_size; (void)ws_size;
    const float* x   = (const float*)d_in[0];
    const float* w0  = (const float*)d_in[1];
    const float* w12 = (const float*)d_in[2];
    const float* lg  = (const float*)d_in[3];
    const float* lb  = (const float*)d_in[4];
    const float* gw  = (const float*)d_in[5];
    float* out = (float*)d_out;

    char* ws = (char*)d_ws;
    size_t off = 0;
    auto alloc = [&](size_t bytes) -> void* {
        void* p = ws + off;
        off += (bytes + 255) & ~(size_t)255;
        return p;
    };
    __bf16* xb  = (__bf16*)alloc((size_t)2 * 64 * 64 * 64 * 32 * 2);
    __bf16* wf0 = (__bf16*)alloc((size_t)27 * 1 * 8 * 32 * 16 * 2);
    __bf16* wf1 = (__bf16*)alloc((size_t)27 * 4 * 8 * 32 * 16 * 2);
    __bf16* wf2 = (__bf16*)alloc((size_t)27 * 4 * 8 * 32 * 16 * 2);
    float*  h0  = (float*)alloc((size_t)2 * 64 * 64 * 64 * 120 * 4);
    __bf16* p0  = (__bf16*)alloc((size_t)2 * 32 * 32 * 32 * 128 * 2);
    float*  h1  = (float*)alloc((size_t)2 * 32 * 32 * 32 * 120 * 4);
    __bf16* p1  = (__bf16*)alloc((size_t)2 * 16 * 16 * 16 * 128 * 2);

    {   // conversions
        size_t tot = (size_t)2 * 64 * 64 * 64 * 32;
        convert_x_kernel<<<(unsigned)((tot + 255) / 256), 256, 0, stream>>>(x, xb, tot);
        size_t t0 = (size_t)27 * 1 * 8 * 32 * 16;
        convert_wfrag_kernel<<<(unsigned)((t0 + 255) / 256), 256, 0, stream>>>(w0, wf0, 8, 1, 120, t0);
        size_t t1 = (size_t)27 * 4 * 8 * 32 * 16;
        convert_wfrag_kernel<<<(unsigned)((t1 + 255) / 256), 256, 0, stream>>>(w12, wf1, 120, 4, 120, t1);
        convert_wfrag_kernel<<<(unsigned)((t1 + 255) / 256), 256, 0, stream>>>(
            w12 + (size_t)27 * 120 * 120, wf2, 120, 4, 120, t1);
    }

    // Stage 0: conv(8->120) + LN + gate
    conv_ln_gate_kernel<32, 1><<<2 * 64 * 64 * 4, 256, 0, stream>>>(
        xb, wf0, lg + 0, lb + 0, gw + 0, h0, 64, 64, 64);
    {
        size_t tot = (size_t)2 * 32 * 32 * 32 * 128;
        avgpool_to_bf16_kernel<<<(unsigned)((tot + 255) / 256), 256, 0, stream>>>(h0, p0, 32, 32, 32, tot);
    }
    // Stage 1: conv(120->120) + LN + gate
    conv_ln_gate_kernel<128, 4><<<2 * 32 * 32 * 2, 256, 0, stream>>>(
        p0, wf1, lg + 120, lb + 120, gw + 32 * 24, h1, 32, 32, 32);
    {
        size_t tot = (size_t)2 * 16 * 16 * 16 * 128;
        avgpool_to_bf16_kernel<<<(unsigned)((tot + 255) / 256), 256, 0, stream>>>(h1, p1, 16, 16, 16, tot);
    }
    // Stage 2: conv(120->120) + LN + gate -> final output
    conv_ln_gate_kernel<128, 4><<<2 * 16 * 16 * 1, 256, 0, stream>>>(
        p1, wf2, lg + 240, lb + 240, gw + 2 * 32 * 24, out, 16, 16, 16);
}
